// TemplateGraphFusionModel_74234214744210
// MI455X (gfx1250) — matrix operand order repaired and checked by reading, not compiled, not run on hardware
//
#include <hip/hip_runtime.h>
#include <hip/hip_bf16.h>

typedef __attribute__((ext_vector_type(16))) __bf16 v16bf;
typedef __attribute__((ext_vector_type(8)))  __bf16 v8bf;
typedef __attribute__((ext_vector_type(8)))  float  v8f;

static __device__ __forceinline__ __bf16 f2bf(float f) {
  union { float f; unsigned u; } v; v.f = f;
  unsigned u = v.u;
  unsigned r = u + 0x7FFFu + ((u >> 16) & 1u);   // round-to-nearest-even
  union { unsigned short s; __bf16 b; } o;
  o.s = (unsigned short)(r >> 16);
  return o.b;
}

__global__ void k_fill(float* __restrict__ p, float v, long n) {
  long i = (long)blockIdx.x * blockDim.x + threadIdx.x;
  if (i < n) p[i] = v;
}

__global__ void k_deg(const int* __restrict__ ei, const float* __restrict__ w,
                      float* __restrict__ deg, int E) {
  int e = blockIdx.x * blockDim.x + threadIdx.x;
  if (e < E) atomicAdd(&deg[ei[E + e]], w[e]);
}

__global__ void k_dinv(float* __restrict__ d, int N) {
  int i = blockIdx.x * blockDim.x + threadIdx.x;
  if (i < N) { float x = d[i]; d[i] = (x > 0.f) ? rsqrtf(x) : 0.f; }
}

__global__ void k_norm(const int* __restrict__ ei, const float* __restrict__ w,
                       const float* __restrict__ dinv, float* __restrict__ norm, int E) {
  int e = blockIdx.x * blockDim.x + threadIdx.x;
  if (e < E) norm[e] = dinv[ei[e]] * w[e] * dinv[ei[E + e]];
}

// Layer 1 transform: h[n,f] = x[n] * W1[f]   (KIN = 1, outer product)
__global__ void k_layer1(const float* __restrict__ x, const float* __restrict__ W1,
                         float* __restrict__ h, int N) {
  long i = (long)blockIdx.x * blockDim.x + threadIdx.x;
  if (i < (long)N * 64) {
    int n = (int)(i >> 6), f = (int)(i & 63);
    h[i] = x[n] * W1[f];
  }
}

// Convert fp32 W[K,F] (row-major) to bf16 TRANSPOSED Wt[F,K]: makes each WMMA
// B-fragment (one column of a 32x16 K-tile) a contiguous 64-byte run.
__global__ void k_cvt_bf_t(const float* __restrict__ src, __bf16* __restrict__ dst,
                           int K, int F) {
  int i = blockIdx.x * blockDim.x + threadIdx.x;
  if (i < K * F) {
    int k = i / F, n = i % F;
    dst[(long)n * K + k] = f2bf(src[i]);
  }
}

// Edge scatter: out[dst] += h[src] * norm[e]   (L2-resident float atomics)
template <int F>
__global__ void k_agg_edges(const int* __restrict__ ei, const float* __restrict__ norm,
                            const float* __restrict__ h, float* __restrict__ out, int E) {
  constexpr int C = F / 4;
  long t = (long)blockIdx.x * blockDim.x + threadIdx.x;
  if (t >= (long)E * C) return;
  int e = (int)(t / C), c = (int)(t % C);
  // Lookahead prefetch of a future edge's gather row (L2-resident, hides latency)
  int e2 = e + 16;
  if (e2 < E) {
    int s2 = ei[e2];
    __builtin_prefetch(h + (long)s2 * F + (long)c * 4, 0, 0);
  }
  int s = ei[e], d = ei[E + e];
  float sc = norm[e];
  const float4 v = *reinterpret_cast<const float4*>(h + (long)s * F + (long)c * 4);
  float* o = out + (long)d * F + (long)c * 4;
  atomicAdd(o + 0, v.x * sc);
  atomicAdd(o + 1, v.y * sc);
  atomicAdd(o + 2, v.z * sc);
  atomicAdd(o + 3, v.w * sc);
}

// Fused epilogue: self-loop term + bias + ReLU, emit bf16 activations
template <int F>
__global__ void k_epilogue(const float* __restrict__ agg, const float* __restrict__ h,
                           const float* __restrict__ dinv, const float* __restrict__ b,
                           __bf16* __restrict__ actbf, int N) {
  long i = (long)blockIdx.x * blockDim.x + threadIdx.x;
  if (i >= (long)N * F) return;
  int n = (int)(i / F), f = (int)(i % F);
  float di = dinv[n];
  float v = agg[i] + h[i] * di * di + b[f];
  actbf[i] = f2bf(fmaxf(v, 0.f));
}

// WMMA GEMM: C[N,KOUT] = A[N,KIN](bf16) @ W[KIN,KOUT], W given TRANSPOSED as
// Wt[KOUT,KIN] bf16. fp32 accumulate. One wave per 16-row tile; A fragments
// register-resident across KOUT tiles. All fragment feeds are aligned b128 loads.
// Fragment layouts per CDNA5 ISA 7.12.2 (16-bit A 16x32, B 32x16, f32 C/D).
template <int KIN, int KOUT>
__global__ void __launch_bounds__(256)
k_gemm_wmma(const __bf16* __restrict__ A, const __bf16* __restrict__ Wt,
            float* __restrict__ C, int N) {
  const int lane = threadIdx.x & 31;
  const int wave = threadIdx.x >> 5;
  const int tiles = N >> 4;                         // N % 16 == 0 (N = 100000)
  const int rt = blockIdx.x * (blockDim.x >> 5) + wave;
  if (rt >= tiles) return;                          // wave-uniform: EXEC stays all-1s
  const int row0 = rt << 4;

  constexpr int NK = KIN / 32;
  v16bf afrag[NK];
  {
    // A 16x32 bf16: lanes 0-15 row M, K {0..7,16..23}; lanes 16-31 row M, K {8..15,24..31}
    // -> two contiguous 8-elem runs per lane per fragment: two b128 loads.
    const int arow = row0 + (lane & 15);
    const int kb = (lane < 16) ? 0 : 8;
    const __bf16* Ar = A + (long)arow * KIN + kb;
#pragma unroll
    for (int kt = 0; kt < NK; ++kt) {
      v8bf lo = *reinterpret_cast<const v8bf*>(Ar + kt * 32);
      v8bf hi = *reinterpret_cast<const v8bf*>(Ar + kt * 32 + 16);
      afrag[kt] = __builtin_shufflevector(lo, hi, 0, 1, 2, 3, 4, 5, 6, 7,
                                          8, 9, 10, 11, 12, 13, 14, 15);
    }
  }
  // B 32x16 bf16: lane holds column N = lane&15; lanes 0-15 K {0..7,16..23}, lanes 16-31
  // K {8..15,24..31} -> two contiguous 8-elem runs per lane in Wt: two b128 loads.
  const int bn  = lane & 15;
  const int bkb = (lane < 16) ? 0 : 8;
#pragma unroll 1
  for (int ct = 0; ct < KOUT / 16; ++ct) {
    const int col0 = ct << 4;
    const __bf16* Wr = Wt + (long)(col0 + bn) * KIN + bkb;
    v8f acc = {};
#pragma unroll
    for (int kt = 0; kt < NK; ++kt) {
      v8bf lo = *reinterpret_cast<const v8bf*>(Wr + kt * 32);
      v8bf hi = *reinterpret_cast<const v8bf*>(Wr + kt * 32 + 16);
      v16bf bfrag = __builtin_shufflevector(lo, hi, 0, 1, 2, 3, 4, 5, 6, 7,
                                            8, 9, 10, 11, 12, 13, 14, 15);
      acc = __builtin_amdgcn_wmma_f32_16x16x32_bf16(
          /*neg_a=*/false, afrag[kt], /*neg_b=*/false, bfrag,
          /*c_mod=*/(short)0, acc, /*reuse_a=*/false, /*reuse_b=*/false);
    }
    // C/D f32 16x16: lanes 0-15 col N=lane rows 0..7; lanes 16-31 col N=lane-16 rows 8..15
    const int crow = row0 + ((lane < 16) ? 0 : 8);
    const int ccol = col0 + bn;
#pragma unroll
    for (int r = 0; r < 8; ++r) C[(long)(crow + r) * KOUT + ccol] = acc[r];
  }
}

__global__ void k_bias_relu(float* __restrict__ h, const float* __restrict__ b,
                            int F, long n) {
  long i = (long)blockIdx.x * blockDim.x + threadIdx.x;
  if (i < n) h[i] = fmaxf(h[i] + b[(int)(i % F)], 0.f);
}

// Head layer 2: per-node 128-dot product (float4 feeds)
__global__ void k_head2(const float* __restrict__ h, const float* __restrict__ w,
                        const float* __restrict__ b, float* __restrict__ out, int N) {
  int n = blockIdx.x * blockDim.x + threadIdx.x;
  if (n >= N) return;
  const float4* hp = reinterpret_cast<const float4*>(h + (long)n * 128);
  const float4* wp = reinterpret_cast<const float4*>(w);
  float s = 0.f;
#pragma unroll 8
  for (int f = 0; f < 32; ++f) {
    float4 a = hp[f], q = wp[f];
    s += a.x * q.x + a.y * q.y + a.z * q.z + a.w * q.w;
  }
  out[n] = s + b[0];
}

extern "C" void kernel_launch(void* const* d_in, const int* in_sizes, int n_in,
                              void* d_out, int out_size, void* d_ws, size_t ws_size,
                              hipStream_t stream) {
  (void)n_in; (void)out_size; (void)ws_size;
  const float* x   = (const float*)d_in[0];
  const int*   ei  = (const int*)d_in[1];     // [2,E] row-major: src then dst
  const float* ea  = (const float*)d_in[2];
  const float* W1  = (const float*)d_in[3];
  const float* b1  = (const float*)d_in[4];
  const float* W2  = (const float*)d_in[5];
  const float* b2  = (const float*)d_in[6];
  const float* W3  = (const float*)d_in[7];
  const float* b3  = (const float*)d_in[8];
  const float* Wp1 = (const float*)d_in[9];
  const float* bp1 = (const float*)d_in[10];
  const float* Wp2 = (const float*)d_in[11];
  const float* bp2 = (const float*)d_in[12];
  float* out = (float*)d_out;
  const int N = in_sizes[0];   // 100000
  const int E = in_sizes[2];   // 800000

  // Workspace carve-up (~260 MB): dinv | norm | bufA(h) | bufB(agg) | actbf | weights-bf16
  auto al = [](size_t v) { return (v + 255) & ~(size_t)255; };
  char* ws = (char*)d_ws;
  size_t off = 0;
  float*  dinv  = (float*)(ws + off);  off = al(off + (size_t)N * 4);
  float*  norm  = (float*)(ws + off);  off = al(off + (size_t)E * 4);
  float*  bufA  = (float*)(ws + off);  off = al(off + (size_t)N * 256 * 4);
  float*  bufB  = (float*)(ws + off);  off = al(off + (size_t)N * 256 * 4);
  __bf16* actbf = (__bf16*)(ws + off); off = al(off + (size_t)N * 256 * 2);
  __bf16* W2t   = (__bf16*)(ws + off); off = al(off + (size_t)64 * 128 * 2);
  __bf16* W3t   = (__bf16*)(ws + off); off = al(off + (size_t)128 * 256 * 2);
  __bf16* Wp1t  = (__bf16*)(ws + off); off = al(off + (size_t)256 * 128 * 2);

  const int T = 256;
  auto nb = [&](long n) { return (unsigned)((n + T - 1) / T); };

  // GCN normalization: deg (self-loop=1) -> dinv -> edge norms
  k_fill<<<nb(N), T, 0, stream>>>(dinv, 1.0f, N);
  k_deg<<<nb(E), T, 0, stream>>>(ei, ea, dinv, E);
  k_dinv<<<nb(N), T, 0, stream>>>(dinv, N);
  k_norm<<<nb(E), T, 0, stream>>>(ei, ea, dinv, norm, E);

  // One-time per call: weights -> transposed bf16
  k_cvt_bf_t<<<nb(64 * 128), T, 0, stream>>>(W2, W2t, 64, 128);
  k_cvt_bf_t<<<nb(128 * 256), T, 0, stream>>>(W3, W3t, 128, 256);
  k_cvt_bf_t<<<nb(256 * 128), T, 0, stream>>>(Wp1, Wp1t, 256, 128);

  const int tiles = N / 16;                        // 6250
  const unsigned gg = (unsigned)((tiles + 7) / 8); // 8 waves / block

  // ---- Layer 1 (1 -> 64) ----
  k_layer1<<<nb((long)N * 64), T, 0, stream>>>(x, W1, bufA, N);
  k_fill<<<nb((long)N * 64), T, 0, stream>>>(bufB, 0.f, (long)N * 64);
  k_agg_edges<64><<<nb((long)E * 16), T, 0, stream>>>(ei, norm, bufA, bufB, E);
  k_epilogue<64><<<nb((long)N * 64), T, 0, stream>>>(bufB, bufA, dinv, b1, actbf, N);

  // ---- Layer 2 (64 -> 128) ----
  k_gemm_wmma<64, 128><<<gg, 256, 0, stream>>>(actbf, W2t, bufA, N);
  k_fill<<<nb((long)N * 128), T, 0, stream>>>(bufB, 0.f, (long)N * 128);
  k_agg_edges<128><<<nb((long)E * 32), T, 0, stream>>>(ei, norm, bufA, bufB, E);
  k_epilogue<128><<<nb((long)N * 128), T, 0, stream>>>(bufB, bufA, dinv, b2, actbf, N);

  // ---- Layer 3 (128 -> 256) ----
  k_gemm_wmma<128, 256><<<gg, 256, 0, stream>>>(actbf, W3t, bufA, N);
  k_fill<<<nb((long)N * 256), T, 0, stream>>>(bufB, 0.f, (long)N * 256);
  k_agg_edges<256><<<nb((long)E * 64), T, 0, stream>>>(ei, norm, bufA, bufB, E);
  k_epilogue<256><<<nb((long)N * 256), T, 0, stream>>>(bufB, bufA, dinv, b3, actbf, N);

  // ---- Head: Linear(256,128) + ReLU, Linear(128,1) ----
  k_gemm_wmma<256, 128><<<gg, 256, 0, stream>>>(actbf, Wp1t, bufA, N);
  k_bias_relu<<<nb((long)N * 128), T, 0, stream>>>(bufA, bp1, 128, (long)N * 128);
  k_head2<<<nb(N), T, 0, stream>>>(bufA, Wp2, bp2, out, N);
}
